// Linear4bit_81965155877556
// MI455X (gfx1250) — compile-verified, matrix operand order checked by reading, not run on hardware
//
#include <hip/hip_runtime.h>

typedef __attribute__((ext_vector_type(16))) _Float16 v16h;
typedef __attribute__((ext_vector_type(8)))  _Float16 v8h;
typedef __attribute__((ext_vector_type(4)))  _Float16 v4h;
typedef __attribute__((ext_vector_type(8)))  float    v8f;

#define M_DIM 8192   // B * S = 4 * 2048
#define N_DIM 11008  // OUT  (= 43 * 256)
#define K_DIM 4096   // IN
#define BM 128
#define BN 256
#define BK 32
#define LDT 40       // padded LDS row stride in halves (80B): conflict-free b128
#define GROUP_M 8    // L2-blocking of the launch order

__global__ __launch_bounds__(256)
void linear4bit_wmma_f16(const float* __restrict__ x,
                         const int*   __restrict__ wq,
                         const float* __restrict__ scale_p,
                         const float* __restrict__ zp_p,
                         const float* __restrict__ bias,
                         float*       __restrict__ out)
{
    __shared__ __attribute__((aligned(16))) _Float16 lds_a[2][BM][LDT];
    __shared__ __attribute__((aligned(16))) _Float16 lds_b[2][BN][LDT];

    const int tid  = threadIdx.x;
    const int lane = tid & 31;           // wave32
    const int wid  = tid >> 5;           // 8 waves
    const int wm   = wid >> 2;           // 0..1  (M direction, 64 rows each)
    const int wn   = wid & 3;            // 0..3  (N direction, 64 cols each)

    // ---- grouped block ordering: keep GROUP_M x-panels hot in L2 while
    //      sweeping all N, so weights stream ~8x instead of 64x ----
    const int num_pid_m = M_DIM / BM;    // 64
    const int num_pid_n = N_DIM / BN;    // 43
    const int pid       = blockIdx.x;
    const int group     = GROUP_M * num_pid_n;
    const int gid       = pid / group;
    const int local     = pid - gid * group;
    const int pid_m     = gid * GROUP_M + (local % GROUP_M);
    const int pid_n     = local / GROUP_M;
    const int bm        = pid_m * BM;
    const int bn        = pid_n * BN;

    const float scale = scale_p[0];
    const float nzs   = -zp_p[0] * scale;   // w = fma(q, scale, nzs)

    // ---- global -> LDS staging maps ----
    // A: 128 rows x 32 cols fp32, 2 threads per row (16 floats each)
    const int arow = tid >> 1;           // 0..127
    const int acol = (tid & 1) * 16;     // 0 or 16
    // B: 256 rows x 32 cols int32, 1 thread per row (32 ints each)
    const int brow = tid;                // 0..255

    const float* xg = x  + (size_t)(bm + arow) * K_DIM + acol;
    const int*   wg = wq + (size_t)(bn + brow) * K_DIM;

    v8f acc[4][4];
    #pragma unroll
    for (int i = 0; i < 4; ++i)
        #pragma unroll
        for (int j = 0; j < 4; ++j)
            acc[i][j] = (v8f){0.f,0.f,0.f,0.f,0.f,0.f,0.f,0.f};

    // ---- prologue: stage K-tile 0 into LDS buffer 0 ----
    #pragma unroll
    for (int j = 0; j < 4; ++j) {
        float4 av = ((const float4*)xg)[j];
        v4h ah = { (_Float16)av.x, (_Float16)av.y, (_Float16)av.z, (_Float16)av.w };
        *(v4h*)&lds_a[0][arow][acol + j * 4] = ah;
    }
    #pragma unroll
    for (int j = 0; j < 8; ++j) {
        int4 qv = ((const int4*)wg)[j];
        v4h bh = { (_Float16)__builtin_fmaf((float)qv.x, scale, nzs),
                   (_Float16)__builtin_fmaf((float)qv.y, scale, nzs),
                   (_Float16)__builtin_fmaf((float)qv.z, scale, nzs),
                   (_Float16)__builtin_fmaf((float)qv.w, scale, nzs) };
        *(v4h*)&lds_b[0][brow][j * 4] = bh;
    }
    __syncthreads();

    const int khalf = lane >> 4;         // which K-half this lane holds
    const int lr    = lane & 15;         // row (A) / col (B) within a 16-frag

    const int KT = K_DIM / BK;           // 128 K-tiles
    for (int kt = 0; kt < KT; ++kt) {
        const int  cur      = kt & 1;
        const bool has_next = (kt + 1 < KT);

        // ---- issue global fetch for the next K-tile (overlaps the WMMAs) ----
        float4 ax[4]; int4 qx[8];
        if (has_next) {
            const float* xp = xg + (size_t)(kt + 1) * BK;
            const int*   wp = wg + (size_t)(kt + 1) * BK;
            #pragma unroll
            for (int j = 0; j < 4; ++j) ax[j] = ((const float4*)xp)[j];
            #pragma unroll
            for (int j = 0; j < 8; ++j) qx[j] = ((const int4*)wp)[j];
            if (kt + 2 < KT) {           // CDNA5 global_prefetch_b8, 2 tiles ahead
                __builtin_prefetch(xg + (size_t)(kt + 2) * BK, 0, 1);
                __builtin_prefetch(wg + (size_t)(kt + 2) * BK, 0, 1);
            }
        }

        // ---- load fragments from LDS (two ds_load_b128 per 16x32 fragment) ----
        v16h af[4], bf[4];
        #pragma unroll
        for (int i = 0; i < 4; ++i) {
            const _Float16* p = &lds_a[cur][wm * 64 + i * 16 + lr][khalf * 8];
            v8h lo = *(const v8h*)p;          // K = khalf*8 .. +7
            v8h hi = *(const v8h*)(p + 16);   // K = 16 + khalf*8 .. +7
            af[i] = __builtin_shufflevector(lo, hi, 0,1,2,3,4,5,6,7,8,9,10,11,12,13,14,15);
        }
        #pragma unroll
        for (int j = 0; j < 4; ++j) {
            const _Float16* p = &lds_b[cur][wn * 64 + j * 16 + lr][khalf * 8];
            v8h lo = *(const v8h*)p;
            v8h hi = *(const v8h*)(p + 16);
            bf[j] = __builtin_shufflevector(lo, hi, 0,1,2,3,4,5,6,7,8,9,10,11,12,13,14,15);
        }

        // ---- 16 x v_wmma_f32_16x16x32_f16 ----
        #pragma unroll
        for (int i = 0; i < 4; ++i)
            #pragma unroll
            for (int j = 0; j < 4; ++j)
                acc[i][j] = __builtin_amdgcn_wmma_f32_16x16x32_f16(
                    false, af[i], false, bf[j], (short)0, acc[i][j], false, false);

        // ---- convert + stage next tile into the other LDS buffer ----
        if (has_next) {
            const int nb = cur ^ 1;
            #pragma unroll
            for (int j = 0; j < 4; ++j) {
                float4 av = ax[j];
                v4h ah = { (_Float16)av.x, (_Float16)av.y, (_Float16)av.z, (_Float16)av.w };
                *(v4h*)&lds_a[nb][arow][acol + j * 4] = ah;
            }
            #pragma unroll
            for (int j = 0; j < 8; ++j) {
                int4 qv = qx[j];
                v4h bh = { (_Float16)__builtin_fmaf((float)qv.x, scale, nzs),
                           (_Float16)__builtin_fmaf((float)qv.y, scale, nzs),
                           (_Float16)__builtin_fmaf((float)qv.z, scale, nzs),
                           (_Float16)__builtin_fmaf((float)qv.w, scale, nzs) };
                *(v4h*)&lds_b[nb][brow][j * 4] = bh;
            }
        }
        __syncthreads();
    }

    // ---- epilogue: D layout -> lane l, VGPR v holds (M = v + 8*(l>>4), N = l&15)
    //      non-temporal stores keep the 361MB output stream out of L2 ----
    #pragma unroll
    for (int j = 0; j < 4; ++j) {
        const int col = bn + wn * 64 + j * 16 + (lane & 15);
        const float bj = bias[col];
        #pragma unroll
        for (int i = 0; i < 4; ++i) {
            const int row0 = bm + wm * 64 + i * 16 + (lane >> 4) * 8;
            #pragma unroll
            for (int v = 0; v < 8; ++v)
                __builtin_nontemporal_store(acc[i][j][v] + bj,
                                            &out[(size_t)(row0 + v) * N_DIM + col]);
        }
    }
}

extern "C" void kernel_launch(void* const* d_in, const int* in_sizes, int n_in,
                              void* d_out, int out_size, void* d_ws, size_t ws_size,
                              hipStream_t stream) {
    const float* x     = (const float*)d_in[0];
    const int*   wq    = (const int*)d_in[1];
    const float* scale = (const float*)d_in[2];
    const float* zp    = (const float*)d_in[3];
    const float* bias  = (const float*)d_in[4];
    float*       out   = (float*)d_out;

    const int nblocks = (M_DIM / BM) * (N_DIM / BN);   // 64 * 43 = 2752
    linear4bit_wmma_f16<<<dim3(nblocks), dim3(256), 0, stream>>>(x, wq, scale, zp, bias, out);
}
